// MLLABlock_79362405695750
// MI455X (gfx1250) — matrix operand order repaired and checked by reading, not compiled, not run on hardware
//
#include <hip/hip_runtime.h>
#include <cstdint>
#include <cstddef>

// ---------------------------------------------------------------------------
// MLLA block forward for MI455X (gfx1250).
// Big GEMMs (446 GFLOP) run as bf16 WMMA (v_wmma_f32_16x16x32_bf16, f32 acc),
// with pre-transposed bf16 weights (N x K) so both A and B tiles stream into
// LDS via b128 stores, double-buffered with one barrier per K-step.
// Wave tile 64x64 (16 WMMA per K-step per wave), block tile 128x256.
// Everything else (convs/LN/RoPE/d=64 einsums) is scalar f32: roofline noise.
// ---------------------------------------------------------------------------

typedef unsigned short ushort_t;
typedef __bf16 bf16_t;
typedef bf16_t v16bf __attribute__((ext_vector_type(16)));
typedef float v8f __attribute__((ext_vector_type(8)));

static constexpr int Bsz = 8, Lseq = 2048, Cdim = 1024, Hh = 16, HIDn = 4096;

#define DEVFN static __device__ __forceinline__

DEVFN float bf2f(ushort_t h) { return __uint_as_float(((unsigned)h) << 16); }
DEVFN ushort_t f2bf(float f) {
  unsigned u = __float_as_uint(f);
  unsigned r = (u + 0x7FFFu + ((u >> 16) & 1u)) >> 16;
  return (ushort_t)r;
}
DEVFN float silu_f(float x) { return x / (1.f + __expf(-x)); }
DEVFN float gelu_f(float x) { return 0.5f * x * (1.f + erff(x * 0.70710678118654752f)); }
DEVFN float elup1_f(float x) { return x > 0.f ? x + 1.f : __expf(x); }

// ------------- f32 (KxN) -> bf16 transposed (NxK) weight convert -----------
__global__ __launch_bounds__(256) void k_wtrans_bf16(const float* __restrict__ W,
                                                     ushort_t* __restrict__ Wt,
                                                     int K, int N) {
  __shared__ float tile[32][33];
  int kb = blockIdx.y * 32, nb = blockIdx.x * 32;
  int tx = threadIdx.x & 31, ty = threadIdx.x >> 5;  // 32 x 8
#pragma unroll
  for (int i = 0; i < 32; i += 8)
    tile[ty + i][tx] = W[(size_t)(kb + ty + i) * N + nb + tx];
  __syncthreads();
#pragma unroll
  for (int i = 0; i < 32; i += 8)
    Wt[(size_t)(nb + ty + i) * K + kb + tx] = f2bf(tile[tx][ty + i]);
}

// ------------------------- depthwise conv1d (k=3) --------------------------
template <bool RES, int ACT, bool OUTBF>
__global__ void k_dwconv(const float* __restrict__ x, const float* __restrict__ w,
                         const float* __restrict__ bias, void* __restrict__ outv) {
  int idx = blockIdx.x * 256 + threadIdx.x;  // over B*L*C (16M)
  int c = idx & (Cdim - 1);
  int l = (idx >> 10) & (Lseq - 1);
  float mid = x[idx];
  float left = (l > 0) ? x[idx - Cdim] : 0.f;
  float right = (l < Lseq - 1) ? x[idx + Cdim] : 0.f;
  float y = w[c * 3] * left + w[c * 3 + 1] * mid + w[c * 3 + 2] * right + bias[c];
  if (RES) y += mid;
  if (ACT == 1) y = silu_f(y);
  if (OUTBF) ((ushort_t*)outv)[idx] = f2bf(y);
  else ((float*)outv)[idx] = y;
}

// ------------------------------- LayerNorm --------------------------------
__global__ void k_layernorm(const float* __restrict__ x, const float* __restrict__ g,
                            const float* __restrict__ b, ushort_t* __restrict__ out) {
  __shared__ float s1[256], s2[256];
  int row = blockIdx.x, t = threadIdx.x;
  const float* xr = x + (size_t)row * Cdim;
  float v[4];
  float sum = 0.f, sq = 0.f;
#pragma unroll
  for (int i = 0; i < 4; i++) {
    v[i] = xr[t + i * 256];
    sum += v[i];
    sq += v[i] * v[i];
  }
  s1[t] = sum; s2[t] = sq;
  __syncthreads();
  for (int o = 128; o > 0; o >>= 1) {
    if (t < o) { s1[t] += s1[t + o]; s2[t] += s2[t + o]; }
    __syncthreads();
  }
  float mu = s1[0] * (1.f / Cdim);
  float var = s2[0] * (1.f / Cdim) - mu * mu;
  float inv = rsqrtf(var + 1e-5f);
  ushort_t* orow = out + (size_t)row * Cdim;
#pragma unroll
  for (int i = 0; i < 4; i++) {
    int c = t + i * 256;
    orow[c] = f2bf((v[i] - mu) * inv * g[c] + b[c]);
  }
}

// --------------------------- bf16 WMMA GEMM --------------------------------
// D = epilogue(A(MxK,bf16) @ Wt^T + bias [+ res]); Wt is N x K bf16.
// Block tile 128(M) x 256(N), BK=32; 256 threads = 8 waves as 2(M) x 4(N);
// each wave: 64x64 = 4x4 frags of 16x16 -> 16 v_wmma per K-step.
// Double-buffered LDS, one barrier per K-step.
// EPI: 0=none 1=silu 2=gelu(exact)
template <int EPI, bool RES, bool OUTBF>
__global__ __launch_bounds__(256) void k_gemm_bf16(
    const ushort_t* __restrict__ A, const ushort_t* __restrict__ Wt,
    const float* __restrict__ bias, const float* __restrict__ res,
    void* __restrict__ outv, int M, int N, int K) {
  // row stride 40 halves (80B, 16B-aligned) to dodge bank conflicts
  __shared__ __align__(16) ushort_t As[2][128 * 40];
  __shared__ __align__(16) ushort_t Bs[2][256 * 40];
  int tid = threadIdx.x;
  int lane = tid & 31, wid = tid >> 5;
  int wm = wid >> 2, wn = wid & 3;    // 2 waves along M, 4 along N
  int g = lane >> 4, lr = lane & 15;  // WMMA lane group / index within group
  int bm = blockIdx.y * 128, bn = blockIdx.x * 256;

  v8f acc[4][4];
#pragma unroll
  for (int i = 0; i < 4; i++)
#pragma unroll
    for (int j = 0; j < 4; j++)
#pragma unroll
      for (int r = 0; r < 8; r++) acc[i][j][r] = 0.f;

  int ar = tid >> 1, ac = (tid & 1) << 4;  // A tile: 128 rows x 32 halves
  const ushort_t* Aptr = A + (size_t)(bm + ar) * K + ac;
  const ushort_t* Bptr = Wt + (size_t)(bn + tid) * K;  // B tile: 256 rows x 32 halves

  // prologue: stage K-tile 0 into buffer 0
  {
    uint4 a0 = *(const uint4*)(Aptr);
    uint4 a1 = *(const uint4*)(Aptr + 8);
    uint4 b0 = *(const uint4*)(Bptr);
    uint4 b1 = *(const uint4*)(Bptr + 8);
    uint4 b2 = *(const uint4*)(Bptr + 16);
    uint4 b3 = *(const uint4*)(Bptr + 24);
    *(uint4*)&As[0][ar * 40 + ac] = a0;
    *(uint4*)&As[0][ar * 40 + ac + 8] = a1;
    *(uint4*)&Bs[0][tid * 40] = b0;
    *(uint4*)&Bs[0][tid * 40 + 8] = b1;
    *(uint4*)&Bs[0][tid * 40 + 16] = b2;
    *(uint4*)&Bs[0][tid * 40 + 24] = b3;
  }
  __syncthreads();

  int buf = 0;
  for (int kt = 0; kt < K; kt += 32) {
    bool nxt = (kt + 32) < K;
    uint4 a0, a1, b0, b1, b2, b3;
    if (nxt) {  // issue next-tile global loads before the WMMAs
      const ushort_t* ap = Aptr + kt + 32;
      const ushort_t* bp = Bptr + kt + 32;
      a0 = *(const uint4*)(ap);
      a1 = *(const uint4*)(ap + 8);
      b0 = *(const uint4*)(bp);
      b1 = *(const uint4*)(bp + 8);
      b2 = *(const uint4*)(bp + 16);
      b3 = *(const uint4*)(bp + 24);
      __builtin_prefetch(ap + 32, 0, 0);  // global_prefetch_b8, K-tile after next
      __builtin_prefetch(bp + 32, 0, 0);
    }

    union Frag { v16bf v; uint4 q[2]; } fa[4], fb[4];
    // A frag (16x32): lane-group 0 holds K=0..7,16..23; group 1 K=8..15,24..31
#pragma unroll
    for (int i = 0; i < 4; i++) {
      int rowi = wm * 64 + i * 16 + lr;
      fa[i].q[0] = *(const uint4*)&As[buf][rowi * 40 + g * 8];
      fa[i].q[1] = *(const uint4*)&As[buf][rowi * 40 + g * 8 + 16];
    }
    // B frag (32x16): lane-group 0 holds K=0..7,16..23; group 1 K=8..15,24..31
#pragma unroll
    for (int j = 0; j < 4; j++) {
      int colj = wn * 64 + j * 16 + lr;
      fb[j].q[0] = *(const uint4*)&Bs[buf][colj * 40 + g * 8];
      fb[j].q[1] = *(const uint4*)&Bs[buf][colj * 40 + g * 8 + 16];
    }
#pragma unroll
    for (int i = 0; i < 4; i++)
#pragma unroll
      for (int j = 0; j < 4; j++)
        acc[i][j] = __builtin_amdgcn_wmma_f32_16x16x32_bf16(
            false, fa[i].v, false, fb[j].v, (short)0, acc[i][j], false, false);

    if (nxt) {  // stage next tile into the other buffer
      int nb = buf ^ 1;
      *(uint4*)&As[nb][ar * 40 + ac] = a0;
      *(uint4*)&As[nb][ar * 40 + ac + 8] = a1;
      *(uint4*)&Bs[nb][tid * 40] = b0;
      *(uint4*)&Bs[nb][tid * 40 + 8] = b1;
      *(uint4*)&Bs[nb][tid * 40 + 16] = b2;
      *(uint4*)&Bs[nb][tid * 40 + 24] = b3;
    }
    __syncthreads();
    buf ^= 1;
  }

  // epilogue: C/D layout — VGPR r -> M = g*8 + r, N = lane%16 per 16x16 frag
#pragma unroll
  for (int j = 0; j < 4; j++) {
    int col = bn + wn * 64 + j * 16 + lr;
    float bb = bias[col];
#pragma unroll
    for (int i = 0; i < 4; i++) {
      int row0 = bm + wm * 64 + i * 16 + g * 8;
#pragma unroll
      for (int r = 0; r < 8; r++) {
        float val = acc[i][j][r] + bb;
        if (EPI == 1) val = silu_f(val);
        else if (EPI == 2) val = gelu_f(val);
        size_t off = (size_t)(row0 + r) * N + col;
        if (RES) val += res[off];
        if (OUTBF) ((ushort_t*)outv)[off] = f2bf(val);
        else ((float*)outv)[off] = val;
      }
    }
  }
}

// ------------------- ELU+1 and fixed-angle RoPE on q/k ---------------------
__global__ void k_rope(const ushort_t* __restrict__ qk, ushort_t* __restrict__ qe,
                       ushort_t* __restrict__ qr, ushort_t* __restrict__ kr) {
  int row = blockIdx.x;  // B*L
  int t = threadIdx.x;
  const ushort_t* src = qk + (size_t)row * (2 * Cdim);
  size_t obase = (size_t)row * Cdim;
#pragma unroll
  for (int i = 0; i < 2; i++) {
    int p = t + i * 256;  // pair index 0..511
    float th = __expf(-(float)p * (9.210340371976184f / 512.f));  // 10000^{-p/512}
    float ct = __cosf(th), st = __sinf(th);
    float q0 = elup1_f(bf2f(src[2 * p]));
    float q1 = elup1_f(bf2f(src[2 * p + 1]));
    float k0 = elup1_f(bf2f(src[Cdim + 2 * p]));
    float k1 = elup1_f(bf2f(src[Cdim + 2 * p + 1]));
    qe[obase + 2 * p] = f2bf(q0);
    qe[obase + 2 * p + 1] = f2bf(q1);
    qr[obase + 2 * p] = f2bf(q0 * ct - q1 * st);
    qr[obase + 2 * p + 1] = f2bf(q0 * st + q1 * ct);
    kr[obase + 2 * p] = f2bf(k0 * ct - k1 * st);
    kr[obase + 2 * p + 1] = f2bf(k0 * st + k1 * ct);
  }
}

// ------------------------- k_mean over sequence ----------------------------
__global__ void k_kmean(const ushort_t* __restrict__ qk, float* __restrict__ km) {
  __shared__ float s[256];
  int bc = blockIdx.x;  // b*C + c
  int b = bc >> 10, c = bc & (Cdim - 1);
  int t = threadIdx.x;
  float sum = 0.f;
#pragma unroll
  for (int i = 0; i < Lseq / 256; i++) {
    int n = t + i * 256;
    sum += elup1_f(bf2f(qk[((size_t)(b * Lseq + n)) * (2 * Cdim) + Cdim + c]));
  }
  s[t] = sum;
  __syncthreads();
  for (int o = 128; o > 0; o >>= 1) {
    if (t < o) s[t] += s[t + o];
    __syncthreads();
  }
  if (t == 0) km[bc] = s[0] * (1.f / Lseq);
}

// ---------------- kv[b,h] = (1/n) * k_rope^T @ v  (64x64) ------------------
__global__ __launch_bounds__(256) void k_kv(const ushort_t* __restrict__ kr,
                                            const ushort_t* __restrict__ v,
                                            float* __restrict__ kv) {
  __shared__ __align__(16) ushort_t ks[64 * 64];
  __shared__ __align__(16) ushort_t vs[64 * 64];
  int bh = blockIdx.x;
  int b = bh >> 4, h = bh & 15;
  int t = threadIdx.x;
  int e = t & 63, db = (t >> 6) * 16;  // thread owns kv[db..db+15][e]
  float acc[16];
#pragma unroll
  for (int i = 0; i < 16; i++) acc[i] = 0.f;
  int rr = t >> 2, c0 = (t & 3) * 16;
  for (int nt = 0; nt < Lseq; nt += 64) {
    size_t goff = ((size_t)(b * Lseq + nt + rr)) * Cdim + h * 64 + c0;
    *(uint4*)&ks[rr * 64 + c0] = *(const uint4*)&kr[goff];
    *(uint4*)&ks[rr * 64 + c0 + 8] = *(const uint4*)&kr[goff + 8];
    *(uint4*)&vs[rr * 64 + c0] = *(const uint4*)&v[goff];
    *(uint4*)&vs[rr * 64 + c0 + 8] = *(const uint4*)&v[goff + 8];
    __syncthreads();
    for (int nl = 0; nl < 64; nl++) {
      float vv = bf2f(vs[nl * 64 + e]);
#pragma unroll
      for (int i = 0; i < 16; i++) acc[i] += bf2f(ks[nl * 64 + db + i]) * vv;
    }
    __syncthreads();
  }
#pragma unroll
  for (int i = 0; i < 16; i++)
    kv[(size_t)bh * 4096 + (size_t)(db + i) * 64 + e] = acc[i] * (1.f / Lseq);
}

// ------ fused: z, q_rope@kv, lepe depthwise conv, act_res gating -----------
__global__ __launch_bounds__(256) void k_attn_gated(
    const ushort_t* __restrict__ qr, const ushort_t* __restrict__ qe,
    const ushort_t* __restrict__ v, const ushort_t* __restrict__ actres,
    const float* __restrict__ kv, const float* __restrict__ km,
    const float* __restrict__ lw, const float* __restrict__ lb,
    ushort_t* __restrict__ gated) {
  __shared__ float kvs[64 * 64];
  __shared__ float kms[64];
  int bh = blockIdx.x;
  int b = bh >> 4, h = bh & 15;
  int t = threadIdx.x;
  const float* kvp = kv + (size_t)bh * 4096;
  for (int i = t; i < 4096; i += 256) kvs[i] = kvp[i];
  if (t < 64) kms[t] = km[b * Cdim + h * 64 + t];
  __syncthreads();
  int n = blockIdx.y * 256 + t;
  size_t base = ((size_t)(b * Lseq + n)) * Cdim + h * 64;
  float qreg[64];
  float z = 0.f;
#pragma unroll
  for (int d = 0; d < 64; d++) {
    qreg[d] = bf2f(qr[base + d]);
    z += bf2f(qe[base + d]) * kms[d];
  }
  z = 1.f / (z + 1e-6f);
#pragma unroll 4
  for (int e = 0; e < 64; e++) {
    float a = 0.f;
#pragma unroll
    for (int d = 0; d < 64; d++) a += qreg[d] * kvs[d * 64 + e];
    int c = h * 64 + e;
    float vm = bf2f(v[base + e]);
    float vl = (n > 0) ? bf2f(v[base - Cdim + e]) : 0.f;
    float vr = (n < Lseq - 1) ? bf2f(v[base + Cdim + e]) : 0.f;
    float lep = lw[c * 3] * vl + lw[c * 3 + 1] * vm + lw[c * 3 + 2] * vr + lb[c];
    float tot = a * z + lep;
    gated[base + e] = f2bf(tot * bf2f(actres[base + e]));
  }
}

// ---------------------------------------------------------------------------
extern "C" void kernel_launch(void* const* d_in, const int* in_sizes, int n_in,
                              void* d_out, int out_size, void* d_ws, size_t ws_size,
                              hipStream_t stream) {
  (void)in_sizes; (void)n_in; (void)out_size; (void)ws_size;
  const float* x      = (const float*)d_in[0];
  const float* cpe1_w = (const float*)d_in[1];
  const float* cpe1_b = (const float*)d_in[2];
  const float* n1g    = (const float*)d_in[3];
  const float* n1b    = (const float*)d_in[4];
  const float* in_w   = (const float*)d_in[5];
  const float* in_b   = (const float*)d_in[6];
  const float* actp_w = (const float*)d_in[7];
  const float* actp_b = (const float*)d_in[8];
  const float* dwc_w  = (const float*)d_in[9];
  const float* dwc_b  = (const float*)d_in[10];
  const float* qk_w   = (const float*)d_in[11];
  const float* qk_b   = (const float*)d_in[12];
  const float* lepe_w = (const float*)d_in[13];
  const float* lepe_b = (const float*)d_in[14];
  const float* out_w  = (const float*)d_in[15];
  const float* out_b  = (const float*)d_in[16];
  const float* cpe2_w = (const float*)d_in[17];
  const float* cpe2_b = (const float*)d_in[18];
  const float* n2g    = (const float*)d_in[19];
  const float* n2b    = (const float*)d_in[20];
  const float* fc1_w  = (const float*)d_in[21];
  const float* fc1_b  = (const float*)d_in[22];
  const float* fc2_w  = (const float*)d_in[23];
  const float* fc2_b  = (const float*)d_in[24];

  const int NT = Bsz * Lseq;  // 16384 token rows
  char* ws = (char*)d_ws;
  size_t off = 0;
  auto alloc = [&](size_t bytes) -> void* {
    void* p = ws + off;
    off += (bytes + 255) & ~(size_t)255;
    return p;
  };

  // bf16 *transposed* weight copies (N x K), L2-resident (~26 MB total)
  ushort_t* wb_actp = (ushort_t*)alloc((size_t)Cdim * Cdim * 2);
  ushort_t* wb_in   = (ushort_t*)alloc((size_t)Cdim * Cdim * 2);
  ushort_t* wb_qk   = (ushort_t*)alloc((size_t)Cdim * 2 * Cdim * 2);
  ushort_t* wb_out  = (ushort_t*)alloc((size_t)Cdim * Cdim * 2);
  ushort_t* wb_fc1  = (ushort_t*)alloc((size_t)Cdim * HIDn * 2);
  ushort_t* wb_fc2  = (ushort_t*)alloc((size_t)HIDn * Cdim * 2);
  // activations
  float*    x1    = (float*)alloc((size_t)NT * Cdim * 4);        // x + cpe1 (shortcut)
  ushort_t* xn    = (ushort_t*)alloc((size_t)NT * Cdim * 2);     // LN1 out
  ushort_t* actre = (ushort_t*)alloc((size_t)NT * Cdim * 2);     // silu(xn@actp)
  float*    xi0   = (float*)alloc((size_t)NT * Cdim * 4);        // xn@in_w (pre dwc)
  ushort_t* xi    = (ushort_t*)alloc((size_t)NT * Cdim * 2);     // silu(dwc(xi0)) = v
  ushort_t* qkb   = (ushort_t*)alloc((size_t)NT * 2 * Cdim * 2); // xi@qk_w
  ushort_t* qe    = (ushort_t*)alloc((size_t)NT * Cdim * 2);     // elu(q)+1 (pre-rope)
  ushort_t* qrp   = (ushort_t*)alloc((size_t)NT * Cdim * 2);     // rope(q)
  ushort_t* krp   = (ushort_t*)alloc((size_t)NT * Cdim * 2);     // rope(k)
  float*    kmean = (float*)alloc((size_t)Bsz * Cdim * 4);
  float*    kvbuf = (float*)alloc((size_t)Bsz * Hh * 64 * 64 * 4);
  ushort_t* gated = (ushort_t*)alloc((size_t)NT * Cdim * 2);     // (attn+lepe)*act_res
  float*    x2    = (float*)alloc((size_t)NT * Cdim * 4);        // shortcut + gated@out_w
  float*    x3    = (float*)alloc((size_t)NT * Cdim * 4);        // x2 + cpe2
  ushort_t* xn2   = (ushort_t*)alloc((size_t)NT * Cdim * 2);     // LN2 out
  ushort_t* h1    = (ushort_t*)alloc((size_t)NT * HIDn * 2);     // gelu(xn2@fc1)

  // 1) weight conversion + transpose: Wt[n][k] = bf16(W[k][n])
  k_wtrans_bf16<<<dim3(Cdim / 32, Cdim / 32), 256, 0, stream>>>(actp_w, wb_actp, Cdim, Cdim);
  k_wtrans_bf16<<<dim3(Cdim / 32, Cdim / 32), 256, 0, stream>>>(in_w, wb_in, Cdim, Cdim);
  k_wtrans_bf16<<<dim3(2 * Cdim / 32, Cdim / 32), 256, 0, stream>>>(qk_w, wb_qk, Cdim, 2 * Cdim);
  k_wtrans_bf16<<<dim3(Cdim / 32, Cdim / 32), 256, 0, stream>>>(out_w, wb_out, Cdim, Cdim);
  k_wtrans_bf16<<<dim3(HIDn / 32, Cdim / 32), 256, 0, stream>>>(fc1_w, wb_fc1, Cdim, HIDn);
  k_wtrans_bf16<<<dim3(Cdim / 32, HIDn / 32), 256, 0, stream>>>(fc2_w, wb_fc2, HIDn, Cdim);

  const int ELEM_BLOCKS = NT * Cdim / 256;  // 65536
  // 2) cpe1 residual
  k_dwconv<true, 0, false><<<ELEM_BLOCKS, 256, 0, stream>>>(x, cpe1_w, cpe1_b, x1);
  // 3) LN1
  k_layernorm<<<NT, 256, 0, stream>>>(x1, n1g, n1b, xn);
  // 4) act_res = silu(xn@actp), xi0 = xn@in_w
  dim3 g1k(Cdim / 256, NT / 128);
  k_gemm_bf16<1, false, true><<<g1k, 256, 0, stream>>>(xn, wb_actp, actp_b, nullptr, actre, NT, Cdim, Cdim);
  k_gemm_bf16<0, false, false><<<g1k, 256, 0, stream>>>(xn, wb_in, in_b, nullptr, xi0, NT, Cdim, Cdim);
  // 5) xi = silu(dwc(xi0))
  k_dwconv<false, 1, true><<<ELEM_BLOCKS, 256, 0, stream>>>(xi0, dwc_w, dwc_b, xi);
  // 6) qk = xi@qk_w
  dim3 g3k(2 * Cdim / 256, NT / 128);
  k_gemm_bf16<0, false, true><<<g3k, 256, 0, stream>>>(xi, wb_qk, qk_b, nullptr, qkb, NT, 2 * Cdim, Cdim);
  // 7) elu+1 + rope
  k_rope<<<NT, 256, 0, stream>>>(qkb, qe, qrp, krp);
  // 8) k_mean
  k_kmean<<<Bsz * Cdim, 256, 0, stream>>>(qkb, kmean);
  // 9) kv per (b,h)
  k_kv<<<Bsz * Hh, 256, 0, stream>>>(krp, xi, kvbuf);
  // 10) attn core + lepe + gating
  dim3 gak(Bsz * Hh, Lseq / 256);
  k_attn_gated<<<gak, 256, 0, stream>>>(qrp, qe, xi, actre, kvbuf, kmean, lepe_w, lepe_b, gated);
  // 11) x2 = shortcut + gated@out_w + out_b
  k_gemm_bf16<0, true, false><<<g1k, 256, 0, stream>>>(gated, wb_out, out_b, x1, x2, NT, Cdim, Cdim);
  // 12) cpe2 residual
  k_dwconv<true, 0, false><<<ELEM_BLOCKS, 256, 0, stream>>>(x2, cpe2_w, cpe2_b, x3);
  // 13) LN2
  k_layernorm<<<NT, 256, 0, stream>>>(x3, n2g, n2b, xn2);
  // 14) h1 = gelu(xn2@fc1)
  dim3 g5k(HIDn / 256, NT / 128);
  k_gemm_bf16<2, false, true><<<g5k, 256, 0, stream>>>(xn2, wb_fc1, fc1_b, nullptr, h1, NT, HIDn, Cdim);
  // 15) out = x3 + h1@fc2 + fc2_b
  k_gemm_bf16<0, true, false><<<g1k, 256, 0, stream>>>(h1, wb_fc2, fc2_b, x3, d_out, NT, Cdim, HIDn);
}
